// Custom_Model_Linear_Cat_37830071943740
// MI455X (gfx1250) — 8-bit matrix operand order repaired and checked by reading, compiled, not run on hardware
//
#include <hip/hip_runtime.h>
#include <stdint.h>

#define M_DIM 8192
#define K_DIM 2048
#define N_DIM 2048

typedef __attribute__((ext_vector_type(8))) int v8i;

__device__ __forceinline__ v8i load_frag_a(const int8_t* __restrict__ p) {
  v8i a;
  int4 d0 = *reinterpret_cast<const int4*>(p);
  int4 d1 = *reinterpret_cast<const int4*>(p + 32);
  a[0] = d0.x; a[1] = d0.y; a[2] = d0.z; a[3] = d0.w;
  a[4] = d1.x; a[5] = d1.y; a[6] = d1.z; a[7] = d1.w;
  return a;
}

__device__ __forceinline__ v8i load_frag_b(const int8_t* __restrict__ p) {
  v8i b;
  int4 d0 = *reinterpret_cast<const int4*>(p);
  int4 d1 = *reinterpret_cast<const int4*>(p + 32);
  b[0] = d0.x; b[1] = d0.y; b[2] = d0.z; b[3] = d0.w;
  b[4] = d1.x; b[5] = d1.y; b[6] = d1.z; b[7] = d1.w;
  return b;
}

__global__ __launch_bounds__(256) void pack_inp_kernel(
    const int* __restrict__ inp, int8_t* __restrict__ Apack,
    float* __restrict__ rowsumf) {
  __shared__ int red[256];
  const int row = (int)blockIdx.x;
  const int t   = (int)threadIdx.x;
  const int* __restrict__ src = inp + (size_t)row * K_DIM;
  int8_t* __restrict__ dst    = Apack + (size_t)row * K_DIM;

  const int base = t * 8;
  int sum = 0, w0 = 0, w1 = 0;
#pragma unroll
  for (int j = 0; j < 4; ++j) {
    int v = src[base + j];
    sum += v;
    w0 |= (v & 0xFF) << (8 * j);
  }
#pragma unroll
  for (int j = 0; j < 4; ++j) {
    int v = src[base + 4 + j];
    sum += v;
    w1 |= (v & 0xFF) << (8 * j);
  }
  int2 packed; packed.x = w0; packed.y = w1;
  reinterpret_cast<int2*>(dst)[t] = packed;

  red[t] = sum;
#pragma unroll
  for (int s = 128; s > 0; s >>= 1) {
    __syncthreads();
    if (t < s) red[t] += red[t + s];
  }
  __syncthreads();
  if (t == 0) rowsumf[row] = (float)red[0];
}

__global__ __launch_bounds__(256) void pack_w_kernel(
    const int* __restrict__ weight, const float* __restrict__ bias,
    const float* __restrict__ inp_scales, const int* __restrict__ inp_zp,
    const float* __restrict__ w_scales, const int* __restrict__ w_zp,
    int8_t* __restrict__ Wpack, float* __restrict__ alpha,
    float* __restrict__ beta, float* __restrict__ gamma) {
  __shared__ int red[256];
  const int row = (int)blockIdx.x;
  const int t   = (int)threadIdx.x;
  const int* __restrict__ src = weight + (size_t)row * K_DIM;
  int8_t* __restrict__ dst    = Wpack + (size_t)row * K_DIM;

  const int base = t * 8;
  int sum = 0, w0 = 0, w1 = 0;
#pragma unroll
  for (int j = 0; j < 4; ++j) {
    int v = src[base + j];
    sum += v;
    w0 |= (v & 0xFF) << (8 * j);
  }
#pragma unroll
  for (int j = 0; j < 4; ++j) {
    int v = src[base + 4 + j];
    sum += v;
    w1 |= (v & 0xFF) << (8 * j);
  }
  int2 packed; packed.x = w0; packed.y = w1;
  reinterpret_cast<int2*>(dst)[t] = packed;

  red[t] = sum;
#pragma unroll
  for (int s = 128; s > 0; s >>= 1) {
    __syncthreads();
    if (t < s) red[t] += red[t + s];
  }
  __syncthreads();
  if (t == 0) {
    const int   sumw = red[0];
    const float sx   = inp_scales[0];
    const int   zx   = inp_zp[0];
    const float a    = sx * w_scales[row];
    const int   zw   = w_zp[row];
    alpha[row] = a;
    gamma[row] = -a * (float)zw;
    beta[row]  = a * (float)(K_DIM * zx * zw - zx * sumw) + bias[row];
  }
}

__global__ __launch_bounds__(256) void qgemm_wmma_kernel(
    const int8_t* __restrict__ Apack, const int8_t* __restrict__ Wpack,
    const float* __restrict__ rowsumf, const float* __restrict__ alpha,
    const float* __restrict__ beta, const float* __restrict__ gamma,
    float* __restrict__ out) {
  const int lane  = (int)(threadIdx.x & 31u);
  const int wave  = (int)(threadIdx.x >> 5u);
  const int l15   = lane & 15;
  const int khalf = lane >> 4;

  const int m0 = (int)blockIdx.y * 128 + (wave & 3) * 32;
  const int n0 = (int)blockIdx.x * 128 + (wave >> 2) * 64;

  v8i acc[2][4];
#pragma unroll
  for (int i = 0; i < 2; ++i)
#pragma unroll
    for (int j = 0; j < 4; ++j)
#pragma unroll
      for (int e = 0; e < 8; ++e) acc[i][j][e] = 0;

  const int8_t* __restrict__ aptr0 =
      Apack + (size_t)(m0 + l15) * K_DIM + khalf * 16;
  const int8_t* __restrict__ aptr1 = aptr0 + (size_t)16 * K_DIM;
  const int8_t* bptr[4];
#pragma unroll
  for (int j = 0; j < 4; ++j)
    bptr[j] = Wpack + (size_t)(n0 + 16 * j + l15) * K_DIM + khalf * 16;

  for (int k = 0; k < K_DIM; k += 64) {
    __builtin_prefetch(aptr0 + k + 64, 0, 0);
    __builtin_prefetch(bptr[0] + k + 64, 0, 0);

    v8i af[2];
    af[0] = load_frag_a(aptr0 + k);
    af[1] = load_frag_a(aptr1 + k);
    v8i bf[4];
#pragma unroll
    for (int j = 0; j < 4; ++j) bf[j] = load_frag_b(bptr[j] + k);

#pragma unroll
    for (int i = 0; i < 2; ++i)
#pragma unroll
      for (int j = 0; j < 4; ++j)
        acc[i][j] = __builtin_amdgcn_wmma_i32_16x16x64_iu8(
             true, af[i],  true, bf[j], acc[i][j],
             false,  false);
  }

#pragma unroll
  for (int i = 0; i < 2; ++i) {
    const int mbase = m0 + i * 16 + khalf * 8;
    const float4 rs0 = *reinterpret_cast<const float4*>(rowsumf + mbase);
    const float4 rs1 = *reinterpret_cast<const float4*>(rowsumf + mbase + 4);
    const float rs[8] = {rs0.x, rs0.y, rs0.z, rs0.w,
                         rs1.x, rs1.y, rs1.z, rs1.w};
#pragma unroll
    for (int j = 0; j < 4; ++j) {
      const int n  = n0 + 16 * j + l15;
      const float al = alpha[n];
      const float ga = gamma[n];
      const float be = beta[n];
#pragma unroll
      for (int g = 0; g < 8; ++g) {
        const float lin = fmaf(al, (float)acc[i][j][g], fmaf(ga, rs[g], be));
        float* o = out + (size_t)(mbase + g) * (size_t)(2 * N_DIM) + n;
        o[0]     = fmaxf(lin, 0.0f);
        o[N_DIM] = lin;
      }
    }
  }
}

extern "C" void kernel_launch(void* const* d_in, const int* in_sizes, int n_in,
                              void* d_out, int out_size, void* d_ws,
                              size_t ws_size, hipStream_t stream) {
  const int*   inp       = (const int*)d_in[0];
  const int*   weight    = (const int*)d_in[1];
  const float* bias      = (const float*)d_in[2];
  const float* inp_s     = (const float*)d_in[3];
  const int*   inp_zp    = (const int*)d_in[4];
  const float* w_s       = (const float*)d_in[5];
  const int*   w_zp      = (const int*)d_in[6];
  float*       out       = (float*)d_out;

  int8_t* Apack   = (int8_t*)d_ws;
  int8_t* Wpack   = Apack + (size_t)M_DIM * K_DIM;
  float*  rowsumf = (float*)(Wpack + (size_t)N_DIM * K_DIM);
  float*  alpha   = rowsumf + M_DIM;
  float*  beta    = alpha + N_DIM;
  float*  gamma   = beta + N_DIM;

  pack_inp_kernel<<<M_DIM, 256, 0, stream>>>(inp, Apack, rowsumf);
  pack_w_kernel<<<N_DIM, 256, 0, stream>>>(weight, bias, inp_s, inp_zp, w_s,
                                           w_zp, Wpack, alpha, beta, gamma);

  dim3 grid(N_DIM / 128, M_DIM / 128);
  qgemm_wmma_kernel<<<grid, 256, 0, stream>>>(Apack, Wpack, rowsumf, alpha,
                                              beta, gamma, out);
}
